// ConvKernel_60009283059903
// MI455X (gfx1250) — matrix operand order repaired and checked by reading, not compiled, not run on hardware
//
#include <hip/hip_runtime.h>
#include <hip/hip_bf16.h>
#include <stdint.h>

// ---------------- problem constants (from reference) ----------------
#define N_NODES 20000
#define WCH     256
#define NEDGE   256000
#define NWELEM  ((size_t)N_NODES * WCH)
#define EPSV    1e-5f

// ---------------- CDNA5 WMMA types ----------------
typedef __attribute__((ext_vector_type(16))) __bf16       v16bf;
typedef __attribute__((ext_vector_type(8)))  float        v8f;
typedef __attribute__((ext_vector_type(4)))  unsigned int u32x4;

union FragBF { u32x4 u[2]; v16bf v; };   // trivially constructible (ext vectors only)

__device__ __forceinline__ v8f wmma_bf16f32(v16bf a, v16bf b, v8f c) {
  // D(f32 16x16) = A(bf16 16x32) * B(bf16 32x16) + C
  return __builtin_amdgcn_wmma_f32_16x16x32_bf16(false, a, false, b, (short)0, c,
                                                 false, false);
}

// A-fragment K mapping (ISA 7.12.2, 16-bit A 16x32):
// lanes 0-15: elems 0-7 -> K0-7,  elems 8-15 -> K16-23
// lanes16-31: elems 0-7 -> K8-15, elems 8-15 -> K24-31
__device__ __forceinline__ int a_kmap(int j, int half) {
  return (j < 8) ? (half * 8 + j) : (16 + half * 8 + (j - 8));
}

// ======================================================================
// Dense GEMM:  C[M,256] = bf16(A)[M,256] @ bf16(W)[256,256]^T + bias
// one wave per 16x16 C tile; K loop of 8 x wmma_f32_16x16x32_bf16
// mode 0: plain   mode 1: *= deg[m]^dexp[n]   mode 2: += exp(spost[n])*xx[m,n]
// ======================================================================
__global__ __launch_bounds__(256)
void gemm_bias(const __bf16* __restrict__ A, const __bf16* __restrict__ W,
               const float* __restrict__ bias, float* __restrict__ C, int M,
               int mode, const float* __restrict__ deg,
               const float* __restrict__ dexp, const float* __restrict__ xx,
               const float* __restrict__ spost)
{
  const int K = WCH;
  int wave  = (int)((blockIdx.x * blockDim.x + threadIdx.x) >> 5);
  int lane  = threadIdx.x & 31;
  int mtile = wave >> 4;          // 16 n-tiles (N=256) per m-tile row
  int ntile = wave & 15;
  if (mtile * 16 >= M) return;
  int half = lane >> 4, r = lane & 15;

  const __bf16* abase = A + (size_t)(mtile * 16 + r) * K;   // A row for this lane
  const __bf16* bbase = W + (size_t)(ntile * 16 + r) * K;   // B: n = lane&15, k = a_kmap(e, half)

  v8f acc = {0.f, 0.f, 0.f, 0.f, 0.f, 0.f, 0.f, 0.f};
  #pragma unroll 4
  for (int k0 = 0; k0 < K; k0 += 32) {
    FragBF fa, fb;
    fa.u[0] = *(const u32x4*)(abase + k0 + half * 8);        // K 0-7   (of this half)
    fa.u[1] = *(const u32x4*)(abase + k0 + 16 + half * 8);   // K 16-23 (of this half)
    fb.u[0] = *(const u32x4*)(bbase + k0 + half * 8);        // K 0-7   (of this half)
    fb.u[1] = *(const u32x4*)(bbase + k0 + 16 + half * 8);   // K 16-23 (of this half)
    acc = wmma_bf16f32(fa.v, fb.v, acc);
  }

  int n = ntile * 16 + r;                   // C: n = lane&15
  float bn = bias[n];
  #pragma unroll
  for (int r2 = 0; r2 < 8; ++r2) {          // C: m = r2 + 8*half
    int m = mtile * 16 + half * 8 + r2;
    float v = acc[r2] + bn;
    if (mode == 1)      v *= __expf(dexp[n] * __logf(deg[m]));       // deg^degree
    else if (mode == 2) v += __expf(spost[n]) * xx[(size_t)m * WCH + n];
    C[(size_t)m * WCH + n] = v;
  }
}

// ======================================================================
// Fused edge stage: gather(h0[src]+h1[dst]) -> per-32ch GroupNorm ->
// (+emb-bag mean for gate path) -> grouped 32x32 gate/value matmuls (WMMA)
// -> relu(gate)*value -> atomic scatter-sum into agg[dst].
// block = 16 edges, wave w = group w (8 groups). 4 WMMAs per wave.
// ======================================================================
__global__ __launch_bounds__(256)
void edge_glin(const float* __restrict__ h0, const float* __restrict__ h1,
               const int* __restrict__ ei,    // [2,E] flattened
               const int* __restrict__ ea,    // [E,3]
               const float* __restrict__ emb, // [bond,256] fp32
               const __bf16* __restrict__ gw, // [8,32,32] bf16
               const __bf16* __restrict__ vw, // [8,32,32] bf16
               float* __restrict__ agg)       // [N,256] (pre-zeroed)
{
  int lane = threadIdx.x & 31;
  int g    = threadIdx.x >> 5;       // group 0..7
  int tile = blockIdx.x;             // E/16 tiles
  int half = lane >> 4, r = lane & 15;
  int e    = tile * 16 + r;          // A rows: m = lane&15 in both halves

  int src = ei[e];
  int dst = ei[NEDGE + e];
  const float* p0 = h0 + (size_t)src * WCH + g * 32;
  const float* p1 = h1 + (size_t)dst * WCH + g * 32;

  float vals[16];
  float s = 0.f, ss = 0.f;
  #pragma unroll
  for (int j = 0; j < 16; ++j) {
    int k = a_kmap(j, half);
    float v = p0[k] + p1[k];
    vals[j] = v; s += v; ss += v * v;
  }
  // the other 16 K-values of this edge live in lane^16
  s  += __shfl_xor(s, 16, 32);
  ss += __shfl_xor(ss, 16, 32);
  float mu   = s * (1.f / 32.f);
  float rstd = rsqrtf(ss * (1.f / 32.f) - mu * mu + EPSV);

  int a0 = ea[e * 3 + 0] * WCH, a1 = ea[e * 3 + 1] * WCH, a2 = ea[e * 3 + 2] * WCH;
  FragBF fav, fag;                       // value path (norm), gate path (norm+emb)
  #pragma unroll
  for (int j = 0; j < 16; ++j) {
    int k  = a_kmap(j, half);
    int c  = g * 32 + k;
    float nv = (vals[j] - mu) * rstd;
    float em = (emb[a0 + c] + emb[a1 + c] + emb[a2 + c]) * (1.f / 3.f);
    fav.v[j] = (__bf16)nv;
    fag.v[j] = (__bf16)(nv + em);
  }

  #pragma unroll
  for (int oh = 0; oh < 2; ++oh) {       // two 16-wide output halves of the 32 outs
    FragBF fbg, fbv;                     // B: n=lane&15 -> out o=oh*16+n, k=a_kmap(e,half)
    const __bf16* gp = gw + g * 1024 + (oh * 16 + r) * 32 + half * 8;
    const __bf16* vp = vw + g * 1024 + (oh * 16 + r) * 32 + half * 8;
    fbg.u[0] = *(const u32x4*)gp; fbg.u[1] = *(const u32x4*)(gp + 16);
    fbv.u[0] = *(const u32x4*)vp; fbv.u[1] = *(const u32x4*)(vp + 16);
    v8f z  = {0.f, 0.f, 0.f, 0.f, 0.f, 0.f, 0.f, 0.f};
    v8f cg = wmma_bf16f32(fag.v, fbg.v, z);
    v8f cv = wmma_bf16f32(fav.v, fbv.v, z);
    #pragma unroll
    for (int r2 = 0; r2 < 8; ++r2) {
      int m = half * 8 + r2;                       // edge row inside tile
      int d = __shfl(dst, m, 32);                  // dst for that edge (held by lane m)
      float o = fmaxf(cg[r2], 0.f) * cv[r2];
      atomicAdd(&agg[(size_t)d * WCH + g * 32 + oh * 16 + r], o);
    }
  }
}

// ======================================================================
// Node-level norm + grouped gate/value (mix block): og = relu(glin(gn(h),gw))*glin(gn(h),vw)
// ======================================================================
__global__ __launch_bounds__(256)
void node_glin(const float* __restrict__ h, const __bf16* __restrict__ gw,
               const __bf16* __restrict__ vw, float* __restrict__ og)
{
  int lane = threadIdx.x & 31;
  int g    = threadIdx.x >> 5;
  int tile = blockIdx.x;             // N/16
  int half = lane >> 4, r = lane & 15;
  const float* p = h + (size_t)(tile * 16 + r) * WCH + g * 32;

  float vals[16];
  float s = 0.f, ss = 0.f;
  #pragma unroll
  for (int j = 0; j < 16; ++j) {
    int k = a_kmap(j, half);
    float v = p[k];
    vals[j] = v; s += v; ss += v * v;
  }
  s  += __shfl_xor(s, 16, 32);
  ss += __shfl_xor(ss, 16, 32);
  float mu   = s * (1.f / 32.f);
  float rstd = rsqrtf(ss * (1.f / 32.f) - mu * mu + EPSV);

  FragBF fa;
  #pragma unroll
  for (int j = 0; j < 16; ++j) fa.v[j] = (__bf16)((vals[j] - mu) * rstd);

  #pragma unroll
  for (int oh = 0; oh < 2; ++oh) {
    FragBF fbg, fbv;
    const __bf16* gp = gw + g * 1024 + (oh * 16 + r) * 32 + half * 8;
    const __bf16* vp = vw + g * 1024 + (oh * 16 + r) * 32 + half * 8;
    fbg.u[0] = *(const u32x4*)gp; fbg.u[1] = *(const u32x4*)(gp + 16);
    fbv.u[0] = *(const u32x4*)vp; fbv.u[1] = *(const u32x4*)(vp + 16);
    v8f z  = {0.f, 0.f, 0.f, 0.f, 0.f, 0.f, 0.f, 0.f};
    v8f cg = wmma_bf16f32(fa.v, fbg.v, z);
    v8f cv = wmma_bf16f32(fa.v, fbv.v, z);
    #pragma unroll
    for (int r2 = 0; r2 < 8; ++r2) {
      int row = tile * 16 + half * 8 + r2;
      og[(size_t)row * WCH + g * 32 + oh * 16 + r] = fmaxf(cg[r2], 0.f) * cv[r2];
    }
  }
}

// ---------------- elementwise helpers (grid-stride) ----------------
__global__ void cvt_f32_bf16(const float* __restrict__ s, __bf16* __restrict__ d, size_t n) {
  size_t i = (size_t)blockIdx.x * blockDim.x + threadIdx.x;
  size_t st = (size_t)gridDim.x * blockDim.x;
  for (; i < n; i += st) d[i] = (__bf16)s[i];
}
__global__ void zero_f32(float* __restrict__ p, size_t n) {
  size_t i = (size_t)blockIdx.x * blockDim.x + threadIdx.x;
  size_t st = (size_t)gridDim.x * blockDim.x;
  for (; i < n; i += st) p[i] = 0.f;
}
__global__ void accum_add(float* __restrict__ a, const float* __restrict__ b, size_t n) {
  size_t i = (size_t)blockIdx.x * blockDim.x + threadIdx.x;
  size_t st = (size_t)gridDim.x * blockDim.x;
  for (; i < n; i += st) a[i] += b[i];
}
__global__ void final_add(const float* __restrict__ a, const float* __restrict__ b,
                          float* __restrict__ o, size_t n) {
  size_t i = (size_t)blockIdx.x * blockDim.x + threadIdx.x;
  size_t st = (size_t)gridDim.x * blockDim.x;
  for (; i < n; i += st) o[i] = a[i] + b[i];
}
// xx = exp(scale_pre[c]) * x_kernel + x_out ; also emit bf16 copy for the GEMM
__global__ void mix_pre(const float* __restrict__ xk, const float* __restrict__ xo,
                        const float* __restrict__ spre, float* __restrict__ xx,
                        __bf16* __restrict__ xb, size_t n) {
  size_t i = (size_t)blockIdx.x * blockDim.x + threadIdx.x;
  size_t st = (size_t)gridDim.x * blockDim.x;
  for (; i < n; i += st) {
    float v = __expf(spre[i & (WCH - 1)]) * xk[i] + xo[i];
    xx[i] = v; xb[i] = (__bf16)v;
  }
}

// ======================================================================
// host orchestration
// ======================================================================
extern "C" void kernel_launch(void* const* d_in, const int* in_sizes, int n_in,
                              void* d_out, int out_size, void* d_ws, size_t ws_size,
                              hipStream_t stream)
{
  (void)in_sizes; (void)n_in; (void)out_size; (void)ws_size;

  // setup_inputs() dict order; params pytree leaves flatten with sorted dict keys:
  // conv layer: degree, emb, gate_w, post_b, post_w, pre0_b, pre0_w, pre1_b, pre1_w, value_w
  // mix layer : gate_w, post_b, post_w, pre_b, pre_w, scale_post, scale_pre, value_w
  const float* X    = (const float*)d_in[0];
  const float* XRES = (const float*)d_in[1];
  #define CONVP(L, i) ((const float*)d_in[2 + (L) * 10 + (i)])
  #define MIXP(m, i)  ((const float*)d_in[92 + (m) * 8 + (i)])
  const int*   EI[3]; const int* EA[3]; const float* ND[3];
  for (int h = 0; h < 3; ++h) {
    EI[h] = (const int*)d_in[108 + h * 3];
    EA[h] = (const int*)d_in[109 + h * 3];
    ND[h] = (const float*)d_in[110 + h * 3];
  }

  // workspace carve-up
  char* ws = (char*)d_ws;
  auto alloc = [&](size_t bytes) -> char* {
    char* p = ws; ws += (bytes + 255) & ~(size_t)255; return p;
  };
  float*  h0      = (float*)alloc(NWELEM * 4);
  float*  h1      = (float*)alloc(NWELEM * 4);
  float*  agg     = (float*)alloc(NWELEM * 4);
  float*  x_hop   = (float*)alloc(NWELEM * 4);
  float*  x_out   = (float*)alloc(NWELEM * 4);
  float*  x_ker   = (float*)alloc(NWELEM * 4);
  float*  xx_mix  = (float*)alloc(NWELEM * 4);
  float*  og      = (float*)alloc(NWELEM * 4);
  __bf16* xb      = (__bf16*)alloc(NWELEM * 2);
  __bf16* wbA     = (__bf16*)alloc(65536 * 2);
  __bf16* wbB     = (__bf16*)alloc(65536 * 2);
  __bf16* wbC     = (__bf16*)alloc(65536 * 2);
  __bf16* wbG     = (__bf16*)alloc(8192 * 2);
  __bf16* wbV     = (__bf16*)alloc(8192 * 2);

  const int EW = 2048, TB = 256;
  auto cvt  = [&](const float* s, __bf16* d, size_t n) {
    cvt_f32_bf16<<<EW, TB, 0, stream>>>(s, d, n);
  };
  auto gemm = [&](const __bf16* A, const __bf16* W, const float* b, float* C, int M,
                  int mode, const float* deg, const float* dexp,
                  const float* xx, const float* sp) {
    gemm_bias<<<M / 8, TB, 0, stream>>>(A, W, b, C, M, mode, deg, dexp, xx, sp);
  };

  // init: x_hop = x_kernel = x ; x_out = 0
  hipMemcpyAsync(x_hop, X, NWELEM * 4, hipMemcpyDeviceToDevice, stream);
  hipMemcpyAsync(x_ker, X, NWELEM * 4, hipMemcpyDeviceToDevice, stream);
  zero_f32<<<EW, TB, 0, stream>>>(x_out, NWELEM);

  for (int k = 0; k < 3; ++k) {
    for (int hp = 0; hp < 3; ++hp) {
      int L = k * 3 + hp;
      // convert activations + this layer's weights to bf16
      cvt(x_hop, xb, NWELEM);
      cvt(CONVP(L, 6), wbA, 65536);   // pre0_w
      cvt(CONVP(L, 8), wbB, 65536);   // pre1_w
      cvt(CONVP(L, 4), wbC, 65536);   // post_w
      cvt(CONVP(L, 2), wbG, 8192);    // gate_w
      cvt(CONVP(L, 9), wbV, 8192);    // value_w
      gemm(xb, wbA, CONVP(L, 5), h0, N_NODES, 0, nullptr, nullptr, nullptr, nullptr);
      gemm(xb, wbB, CONVP(L, 7), h1, N_NODES, 0, nullptr, nullptr, nullptr, nullptr);
      zero_f32<<<EW, TB, 0, stream>>>(agg, NWELEM);
      edge_glin<<<NEDGE / 16, TB, 0, stream>>>(h0, h1, EI[hp], EA[hp],
                                               CONVP(L, 1), wbG, wbV, agg);
      cvt(agg, xb, NWELEM);
      gemm(xb, wbC, CONVP(L, 3), x_hop, N_NODES, 1, ND[hp], CONVP(L, 0),
           nullptr, nullptr);
      accum_add<<<EW, TB, 0, stream>>>(x_out, x_hop, NWELEM);
    }
    if (k < 2) {
      int m = k;
      mix_pre<<<EW, TB, 0, stream>>>(x_ker, x_out, MIXP(m, 6), xx_mix, xb, NWELEM);
      cvt(MIXP(m, 4), wbA, 65536);    // pre_w
      cvt(MIXP(m, 2), wbC, 65536);    // post_w
      cvt(MIXP(m, 0), wbG, 8192);     // gate_w
      cvt(MIXP(m, 7), wbV, 8192);     // value_w
      gemm(xb, wbA, MIXP(m, 3), h0, N_NODES, 0, nullptr, nullptr, nullptr, nullptr);
      node_glin<<<N_NODES / 16, TB, 0, stream>>>(h0, wbG, wbV, og);
      cvt(og, xb, NWELEM);
      // x_kernel = exp(scale_post)*xx + (og @ post_w^T + post_b)
      gemm(xb, wbC, MIXP(m, 1), x_ker, N_NODES, 2, nullptr, nullptr,
           xx_mix, MIXP(m, 5));
      hipMemcpyAsync(x_hop, x_ker, NWELEM * 4, hipMemcpyDeviceToDevice, stream);
      zero_f32<<<EW, TB, 0, stream>>>(x_out, NWELEM);
    }
  }

  final_add<<<EW, TB, 0, stream>>>(x_out, XRES, (float*)d_out, NWELEM);
}
